// Decoder_56143812493482
// MI455X (gfx1250) — matrix operand order repaired and checked by reading, not compiled, not run on hardware
//
#include <hip/hip_runtime.h>
#include <math.h>

// ---------------------------------------------------------------------------
// Decoder step, MI455X (gfx1250, wave32). Compute-bound part (Bahdanau energy
// GEMM, 12.9 G MACs) and pointer GEMM run on v_wmma_f32_16x16x32_bf16 with
// f32 accumulation; all GEMVs are bandwidth-bound coalesced streamers with
// k-split + f32 atomics into bias-seeded accumulators (re-seeded every call).
// ---------------------------------------------------------------------------

typedef __attribute__((ext_vector_type(16))) __bf16           v16bf;
typedef __attribute__((ext_vector_type(16))) unsigned short   v16u;
typedef __attribute__((ext_vector_type(8)))  float            v8f;

#define H2   1024
#define SX   8192
#define SC   4096
#define SEQT 12288
#define VOC  50257
#define EMB  300
#define CH   6
#define PMR  32

__device__ __forceinline__ unsigned short f2bf(float f) {
  union { float f; unsigned int u; } c; c.f = f;
  unsigned int u = c.u;
  unsigned int r = u + 0x7FFFu + ((u >> 16) & 1u);   // round-to-nearest-even
  return (unsigned short)(r >> 16);
}

__device__ __forceinline__ v8f wmma_bf16(v16u a, v16u b, v8f c) {
  return __builtin_amdgcn_wmma_f32_16x16x32_bf16(
      false, __builtin_bit_cast(v16bf, a),
      false, __builtin_bit_cast(v16bf, b),
      (short)0, c, false, false);
}

__device__ __forceinline__ float sigf(float x) { return 1.0f / (1.0f + expf(-x)); }

// ---------------------------------------------------------------------------
// Pack a [1024,1024] f32 weight (row-major, K x N) into bf16 WMMA B-fragment
// layout: tile (nt,kt) of 32x16, record = 32 lanes x 16 bf16 contiguous.
// B 32x16 bf16 layout: lane l holds column N = l&15; K = (l<16?0:8)+(i<8?i:i+8).
// blockIdx.x < 256 -> Wa bottom half; else -> Wp.
// ---------------------------------------------------------------------------
__global__ __launch_bounds__(256) void k_pack(
    const float* __restrict__ Wa, const float* __restrict__ Wp,
    unsigned short* __restrict__ WaB, unsigned short* __restrict__ WpB)
{
  const int which = blockIdx.x >> 8;
  const float* src = which ? Wp : (Wa + (size_t)H2 * H2);   // bottom half of Wa
  unsigned short* dst = which ? WpB : WaB;
  const int w    = ((blockIdx.x & 255) << 3) + (threadIdx.x >> 5); // tile id 0..2047
  const int kt   = w & 31;
  const int nt   = w >> 5;
  const int lane = threadIdx.x & 31;
  const int koff = (lane < 16) ? 0 : 8;
  const int col  = nt * 16 + (lane & 15);
  v16u vals;
#pragma unroll
  for (int i = 0; i < 16; ++i)
    vals[i] = f2bf(src[(size_t)(kt * 32 + koff + (i < 8 ? i : i + 8)) * H2 + col]);
  *(v16u*)(dst + ((size_t)((nt << 5) + kt) * 32 + lane) * 16) = vals;
}

// ---------------------------------------------------------------------------
// Bahdanau energy GEMM: scores[s] = va . tanh(hWa + enc[s] @ Wa_bot), masked.
// One block = one 16-row tile; 8 waves x 8 n-tiles each, k-outer WMMA loop.
// ---------------------------------------------------------------------------
__global__ __launch_bounds__(256) void k_energy(
    const float* __restrict__ encx, const float* __restrict__ encc,
    const unsigned short* __restrict__ WaB,
    const float* __restrict__ hWa, const float* __restrict__ va,
    const unsigned char* __restrict__ mask_x, const unsigned char* __restrict__ mask_c,
    float* __restrict__ scores)
{
  __shared__ float sscore[16];
  const int tid  = threadIdx.x;
  const int wave = tid >> 5;
  const int lane = tid & 31;
  const int m0   = blockIdx.x * 16;

  const float* A;
  int arow0;
  if (m0 < SX) { A = encx; arow0 = m0; } else { A = encc; arow0 = m0 - SX; }

  if (tid < 16) sscore[tid] = 0.0f;
  __syncthreads();

  // A 16x32 bf16 fragment: lane row = l&15, koff = (l<16?0:8),
  // elements i: K = koff + (i<8 ? i : i+8)  -> two contiguous 8-float runs.
  const int row  = arow0 + (lane & 15);
  const int koff = (lane < 16) ? 0 : 8;
  const float* arow = A + (size_t)row * H2 + koff;

  v8f zero = {};
  v8f acc[8];
#pragma unroll
  for (int j = 0; j < 8; ++j) acc[j] = zero;

  for (int kt = 0; kt < 32; ++kt) {
    const float* p = arow + kt * 32;
    v16u af;
#pragma unroll
    for (int i = 0; i < 8; ++i) { af[i] = f2bf(p[i]); af[8 + i] = f2bf(p[16 + i]); }
#pragma unroll
    for (int j = 0; j < 8; ++j) {
      const int nt = wave * 8 + j;
      const v16u bf = *(const v16u*)(WaB + ((size_t)((nt << 5) + kt) * 32 + lane) * 16);
      acc[j] = wmma_bf16(af, bf, acc[j]);
    }
  }

  // C 16x16 f32: lane l holds N = l&15, rows M = v + (l<16?0:8), v=0..7.
  float part[8];
#pragma unroll
  for (int v = 0; v < 8; ++v) part[v] = 0.0f;
#pragma unroll
  for (int j = 0; j < 8; ++j) {
    const int n = (wave * 8 + j) * 16 + (lane & 15);
    const float hb = hWa[n], vn = va[n];
#pragma unroll
    for (int v = 0; v < 8; ++v) part[v] += tanhf(acc[j][v] + hb) * vn;
  }
  const int mbase = (lane < 16) ? 0 : 8;
#pragma unroll
  for (int v = 0; v < 8; ++v) atomicAdd(&sscore[mbase + v], part[v]);
  __syncthreads();

  if (tid < 16) {
    const int grow = m0 + tid;
    float s = sscore[tid];
    const bool mk = (grow < SX) ? (mask_x[grow] != 0) : (mask_c[grow - SX] != 0);
    scores[grow] = mk ? s : -1e10f;
  }
}

// ---------------------------------------------------------------------------
// Pointer GEMM: att32[m] = vv . tanh(hWh + bp + pm[sel*32+m] @ Wp), m=0..31.
// Same WMMA tile routine; grid = 2 blocks of 16 rows.
// ---------------------------------------------------------------------------
__global__ __launch_bounds__(256) void k_ppgemm(
    const float* __restrict__ pm, const unsigned short* __restrict__ WpB,
    const float* __restrict__ hWh, const float* __restrict__ bp,
    const float* __restrict__ vv, const int* __restrict__ selp,
    float* __restrict__ att32)
{
  __shared__ float sscore[16];
  const int tid  = threadIdx.x;
  const int wave = tid >> 5;
  const int lane = tid & 31;
  const int m0   = blockIdx.x * 16;
  const int sel  = *selp;

  if (tid < 16) sscore[tid] = 0.0f;
  __syncthreads();

  const int row  = sel * PMR + m0 + (lane & 15);
  const int koff = (lane < 16) ? 0 : 8;
  const float* arow = pm + (size_t)row * H2 + koff;

  v8f zero = {};
  v8f acc[8];
#pragma unroll
  for (int j = 0; j < 8; ++j) acc[j] = zero;

  for (int kt = 0; kt < 32; ++kt) {
    const float* p = arow + kt * 32;
    v16u af;
#pragma unroll
    for (int i = 0; i < 8; ++i) { af[i] = f2bf(p[i]); af[8 + i] = f2bf(p[16 + i]); }
#pragma unroll
    for (int j = 0; j < 8; ++j) {
      const int nt = wave * 8 + j;
      const v16u bf = *(const v16u*)(WpB + ((size_t)((nt << 5) + kt) * 32 + lane) * 16);
      acc[j] = wmma_bf16(af, bf, acc[j]);
    }
  }

  float part[8];
#pragma unroll
  for (int v = 0; v < 8; ++v) part[v] = 0.0f;
#pragma unroll
  for (int j = 0; j < 8; ++j) {
    const int n = (wave * 8 + j) * 16 + (lane & 15);
    const float hb = hWh[n] + bp[n];
    const float vn = vv[n];
#pragma unroll
    for (int v = 0; v < 8; ++v) part[v] += tanhf(acc[j][v] + hb) * vn;
  }
  const int mbase = (lane < 16) ? 0 : 8;
#pragma unroll
  for (int v = 0; v < 8; ++v) atomicAdd(&sscore[mbase + v], part[v]);
  __syncthreads();

  if (tid < 16) att32[m0 + tid] = sscore[tid];
}

// ---------------------------------------------------------------------------
// Generic GEMV accumulate: acc[n] += sum_{k in chunk} x[k]*W[k,N] (row-major W)
// acc must be pre-seeded with bias. Coalesced across n; k-split via grid.y.
// ---------------------------------------------------------------------------
__global__ __launch_bounds__(256) void k_gemv_acc(
    const float* __restrict__ x, const float* __restrict__ W,
    int K, int N, int kchunk, float* __restrict__ acc)
{
  const int n = blockIdx.x * 256 + threadIdx.x;
  if (n >= N) return;
  const int k0 = blockIdx.y * kchunk;
  const int k1 = min(K, k0 + kchunk);
  float s = 0.0f;
  for (int k = k0; k < k1; ++k) s += x[k] * W[(size_t)k * N + n];
  atomicAdd(&acc[n], s);
}

// Seed all accumulators with biases / zeros (deterministic per call).
__global__ __launch_bounds__(256) void k_seed(
    float* hWa, float* hWh, float* wacc, float* wOut, float* yin,
    float* gates, float* logits,
    const float* __restrict__ ba, const float* __restrict__ bh,
    const float* __restrict__ bg, const float* __restrict__ by,
    const float* __restrict__ bih, const float* __restrict__ bhh,
    const float* __restrict__ bout)
{
  const int id = blockIdx.x * 256 + threadIdx.x;
  if (id < H2)    { hWa[id] = ba[id]; hWh[id] = bh[id]; wOut[id] = bg[id]; }
  if (id < 2*H2)  wacc[id] = 0.0f;
  if (id < EMB)   yin[id] = by[id];
  if (id < 4*H2)  gates[id] = bih[id] + bhh[id];
  if (id < VOC)   logits[id] = bout[id];
}

// Softmax over the two attention segments (in place). grid = 2 blocks.
__global__ __launch_bounds__(256) void k_softmax_seg(float* __restrict__ scores)
{
  __shared__ float red[256];
  const int tid  = threadIdx.x;
  const int base = (blockIdx.x == 0) ? 0 : SX;
  const int n    = (blockIdx.x == 0) ? SX : SC;
  float m = -3.4e38f;
  for (int i = tid; i < n; i += 256) m = fmaxf(m, scores[base + i]);
  red[tid] = m; __syncthreads();
  for (int s = 128; s > 0; s >>= 1) { if (tid < s) red[tid] = fmaxf(red[tid], red[tid + s]); __syncthreads(); }
  const float mx = red[0]; __syncthreads();
  float sum = 0.0f;
  for (int i = tid; i < n; i += 256) sum += expf(scores[base + i] - mx);
  red[tid] = sum; __syncthreads();
  for (int s = 128; s > 0; s >>= 1) { if (tid < s) red[tid] += red[tid + s]; __syncthreads(); }
  const float inv = 1.0f / red[0]; __syncthreads();
  for (int i = tid; i < n; i += 256) scores[base + i] = expf(scores[base + i] - mx) * inv;
}

// weighted_{x,c}[d] = sum_s a[s]*enc[s,d]; 48 blocks stream 50MB (L2-hot).
__global__ __launch_bounds__(256) void k_wsum(
    const float* __restrict__ encx, const float* __restrict__ encc,
    const float* __restrict__ scores, float* __restrict__ wacc)
{
  const int tid = threadIdx.x;
  const float* enc; const float* w; float* out; int s0;
  if (blockIdx.x < 32) { enc = encx; w = scores;      out = wacc;      s0 = blockIdx.x * 256; }
  else                 { enc = encc; w = scores + SX; out = wacc + H2; s0 = (blockIdx.x - 32) * 256; }
  float a0 = 0, a1 = 0, a2 = 0, a3 = 0;
  for (int s = s0; s < s0 + 256; ++s) {
    const float a = w[s];
    const float* r = enc + (size_t)s * H2;
    a0 += a * r[tid];       a1 += a * r[tid + 256];
    a2 += a * r[tid + 512]; a3 += a * r[tid + 768];
  }
  atomicAdd(&out[tid], a0);       atomicAdd(&out[tid + 256], a1);
  atomicAdd(&out[tid + 512], a2); atomicAdd(&out[tid + 768], a3);
}

// Char gate: g = [y_input, h0, weighted] @ Wchar + bchar; nz from pm sums;
// sel = argmax over nz-active chars (softmax is monotone). Also builds xcat.
__global__ __launch_bounds__(256) void k_char(
    const int* __restrict__ idx_p, const float* __restrict__ embed,
    const float* __restrict__ h0, const float* __restrict__ wOut,
    const float* __restrict__ Wchar, const float* __restrict__ bchar,
    const float* __restrict__ pm,
    float* __restrict__ xcat, float* __restrict__ yinput, int* __restrict__ selp)
{
  __shared__ float sg[CH];
  __shared__ float snz[CH];
  const int tid = threadIdx.x;
  const int idx = idx_p[0];
  for (int i = tid; i < EMB; i += 256) { const float v = embed[(size_t)idx * EMB + i]; yinput[i] = v; xcat[i] = v; }
  for (int i = tid; i < H2; i += 256) xcat[EMB + i] = wOut[i];
  if (tid < CH) {
    float s = bchar[tid];
    for (int k = 0; k < EMB; ++k) s += embed[(size_t)idx * EMB + k] * Wchar[k * CH + tid];
    for (int k = 0; k < H2; ++k)  s += h0[k]   * Wchar[(EMB + k) * CH + tid];
    for (int k = 0; k < H2; ++k)  s += wOut[k] * Wchar[(EMB + H2 + k) * CH + tid];
    sg[tid] = s; snz[tid] = 0.0f;
  }
  __syncthreads();
  for (int j = 0; j < CH; ++j) {
    float p = 0.0f;
    for (int i = tid; i < PMR * H2; i += 256) p += pm[(size_t)j * PMR * H2 + i];
    atomicAdd(&snz[j], p);
  }
  __syncthreads();
  if (tid == 0) {
    int sel = 0; float best = -3.4e38f; bool any = false;
    for (int j = 0; j < CH; ++j)
      if (snz[j] != 0.0f) { any = true; if (sg[j] > best) { best = sg[j]; sel = j; } }
    if (!any) sel = 0;
    *selp = sel;
  }
}

// Pointer softmax over 192 slots (160 zero-rows share one score), p_attn,
// and the p_attn_score_sel output (32 values).
__global__ __launch_bounds__(256) void k_pattn(
    const float* __restrict__ pm, const float* __restrict__ hWh,
    const float* __restrict__ bp, const float* __restrict__ vv,
    const int* __restrict__ selp, const float* __restrict__ att32,
    float* __restrict__ pattn, float* __restrict__ xcat, float* __restrict__ dout)
{
  __shared__ float red[256];
  __shared__ float s32[PMR];
  __shared__ float mx_s, den_s;
  const int tid = threadIdx.x;
  float p = 0.0f;
  for (int n = tid; n < H2; n += 256) p += vv[n] * tanhf(hWh[n] + bp[n]);
  red[tid] = p; __syncthreads();
  for (int s = 128; s > 0; s >>= 1) { if (tid < s) red[tid] += red[tid + s]; __syncthreads(); }
  if (tid == 0) {
    const float az = red[0];
    float mx = az;
    for (int i = 0; i < PMR; ++i) mx = fmaxf(mx, att32[i]);
    float den = (float)(CH * PMR - PMR) * expf(az - mx);
    for (int i = 0; i < PMR; ++i) den += expf(att32[i] - mx);
    mx_s = mx; den_s = den;
  }
  __syncthreads();
  if (tid < PMR) {
    const float v = expf(att32[tid] - mx_s) / den_s;
    s32[tid] = v;
    dout[VOC + 3 * H2 + tid] = v;       // p_attn_score_sel
  }
  __syncthreads();
  const int sel = *selp;
  for (int d = tid; d < H2; d += 256) {
    float acc = 0.0f;
    for (int i = 0; i < PMR; ++i) acc += s32[i] * pm[(size_t)(sel * PMR + i) * H2 + d];
    pattn[d] = acc;
    xcat[EMB + H2 + d] = acc;
  }
}

// LSTM cell (torch gate order i,f,g,o) + write h1/c1/weighted outputs.
__global__ __launch_bounds__(256) void k_lstm(
    const float* __restrict__ gates, const float* __restrict__ c0,
    const float* __restrict__ wOut, float* __restrict__ h1ws, float* __restrict__ dout)
{
  for (int n = threadIdx.x; n < H2; n += 256) {
    const float gi = gates[n], gf = gates[n + H2], gg = gates[n + 2 * H2], go = gates[n + 3 * H2];
    const float c1 = sigf(gf) * c0[n] + sigf(gi) * tanhf(gg);
    const float h1 = sigf(go) * tanhf(c1);
    h1ws[n] = h1;
    dout[VOC + n]          = h1;
    dout[VOC + H2 + n]     = c1;
    dout[VOC + 2 * H2 + n] = wOut[n];
  }
}

// Softmax over the vocab logits -> probabilities in d_out[0:VOC].
__global__ __launch_bounds__(1024) void k_softmax_out(
    const float* __restrict__ logits, float* __restrict__ dout)
{
  __shared__ float red[1024];
  const int tid = threadIdx.x;
  float m = -3.4e38f;
  for (int v = tid; v < VOC; v += 1024) m = fmaxf(m, logits[v]);
  red[tid] = m; __syncthreads();
  for (int s = 512; s > 0; s >>= 1) { if (tid < s) red[tid] = fmaxf(red[tid], red[tid + s]); __syncthreads(); }
  const float mx = red[0]; __syncthreads();
  float sum = 0.0f;
  for (int v = tid; v < VOC; v += 1024) sum += expf(logits[v] - mx);
  red[tid] = sum; __syncthreads();
  for (int s = 512; s > 0; s >>= 1) { if (tid < s) red[tid] += red[tid + s]; __syncthreads(); }
  const float inv = 1.0f / red[0]; __syncthreads();
  for (int v = tid; v < VOC; v += 1024) dout[v] = expf(logits[v] - mx) * inv;
}

// ---------------------------------------------------------------------------
extern "C" void kernel_launch(void* const* d_in, const int* in_sizes, int n_in,
                              void* d_out, int out_size, void* d_ws, size_t ws_size,
                              hipStream_t stream) {
  (void)in_sizes; (void)n_in; (void)out_size; (void)ws_size;

  const int*   input_idx = (const int*)  d_in[0];
  const float* encx   = (const float*)d_in[1];
  const float* encc   = (const float*)d_in[2];
  const float* pm     = (const float*)d_in[3];
  const float* h0     = (const float*)d_in[4];
  const float* c0     = (const float*)d_in[5];
  const unsigned char* mask_x = (const unsigned char*)d_in[6];
  const unsigned char* mask_c = (const unsigned char*)d_in[7];
  const float* embed  = (const float*)d_in[8];
  const float* Wa     = (const float*)d_in[9];
  const float* ba     = (const float*)d_in[10];
  const float* va     = (const float*)d_in[11];
  const float* Wg     = (const float*)d_in[12];
  const float* bg     = (const float*)d_in[13];
  const float* Wchar  = (const float*)d_in[14];
  const float* bchar  = (const float*)d_in[15];
  const float* Wp     = (const float*)d_in[16];
  const float* bp     = (const float*)d_in[17];
  const float* Wh     = (const float*)d_in[18];
  const float* bh     = (const float*)d_in[19];
  const float* vv     = (const float*)d_in[20];
  const float* Wy     = (const float*)d_in[21];
  const float* by     = (const float*)d_in[22];
  const float* Wih    = (const float*)d_in[23];
  const float* Whh    = (const float*)d_in[24];
  const float* bih    = (const float*)d_in[25];
  const float* bhh    = (const float*)d_in[26];
  const float* Wout   = (const float*)d_in[27];
  const float* bout   = (const float*)d_in[28];
  float* dout = (float*)d_out;

  // Workspace carve-up (all offsets 256B-aligned, ~4.5 MB total).
  char* wsb = (char*)d_ws;
  size_t o = 0;
  unsigned short* WaB = (unsigned short*)(wsb + o); o += (size_t)H2 * H2 * 2;   // 2 MB
  unsigned short* WpB = (unsigned short*)(wsb + o); o += (size_t)H2 * H2 * 2;   // 2 MB
  float* scores = (float*)(wsb + o); o += (size_t)SEQT * 4;
  float* hWa    = (float*)(wsb + o); o += 4096;
  float* hWh    = (float*)(wsb + o); o += 4096;
  float* wacc   = (float*)(wsb + o); o += 8192;       // [weighted_x | weighted_c]
  float* wOut   = (float*)(wsb + o); o += 4096;       // gated weighted
  int*   selp   = (int*)  (wsb + o); o += 256;
  float* att32  = (float*)(wsb + o); o += 256;
  float* pattn  = (float*)(wsb + o); o += 4096;
  float* yinput = (float*)(wsb + o); o += 1280;
  float* xcat   = (float*)(wsb + o); o += 9472;       // [y_input|weighted|p_attn] 2348
  float* yin    = (float*)(wsb + o); o += 1280;
  float* gates  = (float*)(wsb + o); o += 16384;
  float* h1ws   = (float*)(wsb + o); o += 4096;
  float* logits = (float*)(wsb + o); o += (size_t)VOC * 4;

  // 1. Seed accumulators with biases (every call -> graph-replay safe).
  k_seed<<<dim3((VOC + 255) / 256), 256, 0, stream>>>(
      hWa, hWh, wacc, wOut, yin, gates, logits, ba, bh, bg, by, bih, bhh, bout);

  // 2. Pack Wa_bot and Wp to bf16 fragment layout.
  k_pack<<<512, 256, 0, stream>>>(Wa, Wp, WaB, WpB);

  // 3. hWa = h0 @ Wa_top (+ba seeded); hWh = h0 @ Wh (+bh seeded).
  k_gemv_acc<<<dim3(4, 4), 256, 0, stream>>>(h0, Wa, H2, H2, 256, hWa);
  k_gemv_acc<<<dim3(4, 4), 256, 0, stream>>>(h0, Wh, H2, H2, 256, hWh);

  // 4. Bahdanau energy GEMM (WMMA bf16), masked scores.
  k_energy<<<SEQT / 16, 256, 0, stream>>>(encx, encc, WaB, hWa, va, mask_x, mask_c, scores);

  // 5. Per-segment softmax; 6. weighted sums.
  k_softmax_seg<<<2, 256, 0, stream>>>(scores);
  k_wsum<<<48, 256, 0, stream>>>(encx, encc, scores, wacc);

  // 7. weighted = [wx|wc] @ Wg (+bg seeded).
  k_gemv_acc<<<dim3(4, 8), 256, 0, stream>>>(wacc, Wg, 2 * H2, H2, 256, wOut);

  // 8. char gate + selection; 9. pointer GEMM (WMMA); 10. pointer softmax.
  k_char<<<1, 256, 0, stream>>>(input_idx, embed, h0, wOut, Wchar, bchar, pm, xcat, yinput, selp);
  k_ppgemm<<<2, 256, 0, stream>>>(pm, WpB, hWh, bp, vv, selp, att32);
  k_pattn<<<1, 256, 0, stream>>>(pm, hWh, bp, vv, selp, att32, pattn, xcat, dout);

  // 11. y_in = xcat @ Wy (+by seeded).
  k_gemv_acc<<<dim3(2, 10), 256, 0, stream>>>(xcat, Wy, EMB + 2 * H2, EMB, 256, yin);

  // 12. gates = y_in @ Wih + h0 @ Whh (+bih+bhh seeded); 13. LSTM cell.
  k_gemv_acc<<<dim3(16, 2), 256, 0, stream>>>(yin, Wih, EMB, 4 * H2, 150, gates);
  k_gemv_acc<<<dim3(16, 4), 256, 0, stream>>>(h0, Whh, H2, 4 * H2, 256, gates);
  k_lstm<<<1, 256, 0, stream>>>(gates, c0, wOut, h1ws, dout);

  // 14. vocab logits (206 MB stream, bandwidth floor) + softmax.
  k_gemv_acc<<<dim3((VOC + 255) / 256, 4), 256, 0, stream>>>(h1ws, Wout, H2, VOC, 256, logits);
  k_softmax_out<<<1, 1024, 0, stream>>>(logits, dout);
}
